// AttentionNoPE_40853728920066
// MI455X (gfx1250) — compile-verified, matrix operand order checked by reading, not run on hardware
//
#include <hip/hip_runtime.h>
#include <hip/hip_bf16.h>

// ---------------- types ----------------
typedef __bf16 bf16_t;
typedef __bf16 v16bf __attribute__((ext_vector_type(16)));
typedef __bf16 v4bf  __attribute__((ext_vector_type(4)));
typedef float  v8f   __attribute__((ext_vector_type(8)));
typedef unsigned int u32x4 __attribute__((ext_vector_type(4)));

#define DEV static __device__ __forceinline__

// ---------------- problem constants ----------------
constexpr int CB = 2, CS = 2048, CD = 2048, CH = 32, CKVH = 8, CHD = 64;
// GROUPS = CH/CKVH = 4

// ---------------- WMMA helpers ----------------
union FragU { u32x4 q[2]; v16bf v; };

// Load a 16-bit A/B operand fragment: lane supplies its row pointer `row`
// (row-contiguous in memory); k0 = K-chunk base (multiple of 32); hf = lane>>4.
// Per ISA layout: vgprs 0-3 hold K = hf*8+0..7, vgprs 4-7 hold K = 16+hf*8+0..7.
DEV v16bf load_frag(const bf16_t* row, int k0, int hf) {
  FragU u;
  u.q[0] = *reinterpret_cast<const u32x4*>(row + k0 + hf * 8);
  u.q[1] = *reinterpret_cast<const u32x4*>(row + k0 + 16 + hf * 8);
  return u.v;
}

DEV v8f wmma_bf16(v16bf a, v16bf b, v8f c) {
  // (neg_a, A, neg_b, B, c_mod, C, reuse_a, reuse_b)
  return __builtin_amdgcn_wmma_f32_16x16x32_bf16(false, a, false, b, (short)0, c,
                                                 false, false);
}

DEV v8f splat8(float x) { v8f v = {x, x, x, x, x, x, x, x}; return v; }

// ---------------- f32 -> bf16 conversion ----------------
__global__ __launch_bounds__(256) void k_cvt(const float* __restrict__ src,
                                             bf16_t* __restrict__ dst) {
  size_t i = (size_t)blockIdx.x * blockDim.x + threadIdx.x;
  float4 f = reinterpret_cast<const float4*>(src)[i];
  v4bf o;
  o[0] = (bf16_t)f.x; o[1] = (bf16_t)f.y; o[2] = (bf16_t)f.z; o[3] = (bf16_t)f.w;
  reinterpret_cast<v4bf*>(dst)[i] = o;
}

// ---------------- core GEMM: C[32x64] per wave, C = A @ W^T ----------------
// A: [M,K] bf16 row-major;  W: [N,K] bf16 row-major (nn.Linear weight).
DEV void gemm_acc_32x64(const bf16_t* __restrict__ A, const bf16_t* __restrict__ W,
                        int K, int m0, int n0, int l16, int hf, v8f acc[2][4]) {
  const bf16_t* ar0 = A + (size_t)(m0 + l16) * K;
  const bf16_t* ar1 = ar0 + (size_t)16 * K;
  const bf16_t* wr  = W + (size_t)(n0 + l16) * K;
  for (int k0 = 0; k0 < K; k0 += 32) {
    v16bf a0 = load_frag(ar0, k0, hf);
    v16bf a1 = load_frag(ar1, k0, hf);
    __builtin_prefetch(ar0 + k0 + 128, 0, 1);
    __builtin_prefetch(ar1 + k0 + 128, 0, 1);
#pragma unroll
    for (int nt = 0; nt < 4; ++nt) {
      v16bf bfr = load_frag(wr + (size_t)(nt * 16) * K, k0, hf);
      acc[0][nt] = wmma_bf16(a0, bfr, acc[0][nt]);
      acc[1][nt] = wmma_bf16(a1, bfr, acc[1][nt]);
    }
  }
}

// C-layout decode: vgpr r, lane l -> row = r + 8*(l>>4), col = l&15.

// ---------------- Q projection: -> Qbf [b,h,s,hd] bf16 ----------------
__global__ __launch_bounds__(256) void k_proj_q(const bf16_t* __restrict__ A,
                                                const bf16_t* __restrict__ W,
                                                bf16_t* __restrict__ Qb) {
  const int lane = threadIdx.x & 31, l16 = lane & 15, hf = lane >> 4;
  const int wid = blockIdx.x * 8 + (threadIdx.x >> 5);
  const int MT = (CB * CS) / 32;             // 128
  const int m0 = (wid % MT) * 32, n0 = (wid / MT) * 64;
  v8f acc[2][4];
#pragma unroll
  for (int i = 0; i < 2; ++i)
#pragma unroll
    for (int nt = 0; nt < 4; ++nt) acc[i][nt] = splat8(0.0f);
  gemm_acc_32x64(A, W, CD, m0, n0, l16, hf, acc);
#pragma unroll
  for (int i = 0; i < 2; ++i)
#pragma unroll
    for (int nt = 0; nt < 4; ++nt)
#pragma unroll
      for (int r = 0; r < 8; ++r) {
        int m = m0 + i * 16 + r + 8 * hf;     // = b*S + s
        int n = n0 + nt * 16 + l16;           // = h*64 + d
        int bb = m >> 11, s = m & 2047, hh = n >> 6, d = n & 63;
        Qb[(((size_t)bb * CH + hh) * CS + s) * CHD + d] = (bf16_t)acc[i][nt][r];
      }
}

// ---------------- K/V projection ----------------
// vmode==0: outb = Kbf [b,kvh,s,hd];  vmode==1: outb = Vt [b,kvh,hd,s].
// cache: f32 [b,kvh,s,hd] (reference kv-cache output).
__global__ __launch_bounds__(256) void k_proj_kv(const bf16_t* __restrict__ A,
                                                 const bf16_t* __restrict__ W,
                                                 bf16_t* __restrict__ outb,
                                                 float* __restrict__ cache,
                                                 int vmode) {
  const int lane = threadIdx.x & 31, l16 = lane & 15, hf = lane >> 4;
  const int wid = blockIdx.x * 8 + (threadIdx.x >> 5);
  const int MT = (CB * CS) / 32;             // 128
  const int m0 = (wid % MT) * 32, n0 = (wid / MT) * 64;
  v8f acc[2][4];
#pragma unroll
  for (int i = 0; i < 2; ++i)
#pragma unroll
    for (int nt = 0; nt < 4; ++nt) acc[i][nt] = splat8(0.0f);
  gemm_acc_32x64(A, W, CD, m0, n0, l16, hf, acc);
#pragma unroll
  for (int i = 0; i < 2; ++i)
#pragma unroll
    for (int nt = 0; nt < 4; ++nt)
#pragma unroll
      for (int r = 0; r < 8; ++r) {
        int m = m0 + i * 16 + r + 8 * hf;     // = b*S + s
        int n = n0 + nt * 16 + l16;           // = kvh*64 + d
        int bb = m >> 11, s = m & 2047, kvh = n >> 6, d = n & 63;
        float val = acc[i][nt][r];
        size_t cidx = (((size_t)bb * CKVH + kvh) * CS + s) * CHD + d;
        cache[cidx] = val;
        if (vmode)
          outb[(((size_t)bb * CKVH + kvh) * CHD + d) * CS + s] = (bf16_t)val;
        else
          outb[cidx] = (bf16_t)val;
      }
}

// ---------------- O projection: f32 out ----------------
__global__ __launch_bounds__(256) void k_proj_o(const bf16_t* __restrict__ A,
                                                const bf16_t* __restrict__ W,
                                                float* __restrict__ out) {
  const int lane = threadIdx.x & 31, l16 = lane & 15, hf = lane >> 4;
  const int wid = blockIdx.x * 8 + (threadIdx.x >> 5);
  const int MT = (CB * CS) / 32;
  const int m0 = (wid % MT) * 32, n0 = (wid / MT) * 64;
  v8f acc[2][4];
#pragma unroll
  for (int i = 0; i < 2; ++i)
#pragma unroll
    for (int nt = 0; nt < 4; ++nt) acc[i][nt] = splat8(0.0f);
  gemm_acc_32x64(A, W, CD, m0, n0, l16, hf, acc);
#pragma unroll
  for (int i = 0; i < 2; ++i)
#pragma unroll
    for (int nt = 0; nt < 4; ++nt)
#pragma unroll
      for (int r = 0; r < 8; ++r) {
        int m = m0 + i * 16 + r + 8 * hf;
        int n = n0 + nt * 16 + l16;
        out[(size_t)m * CD + n] = acc[i][nt][r];
      }
}

// ---------------- flash attention (causal, GQA), one wave per (b,h,qtile) ----
__global__ __launch_bounds__(256) void k_attn(const bf16_t* __restrict__ Qb,
                                              const bf16_t* __restrict__ Kb,
                                              const bf16_t* __restrict__ Vt,
                                              bf16_t* __restrict__ Ob) {
  __shared__ bf16_t pbuf[8 * 16 * 40];   // per-wave 16x32 P tile, 40-elt (80B) row stride
  const int lane = threadIdx.x & 31, l16 = lane & 15, hf = lane >> 4;
  const int wib = threadIdx.x >> 5;
  const int wid = blockIdx.x * 8 + wib;
  const int qt = wid & 127;               // q tile (16 rows)
  const int hh = (wid >> 7) & 31;         // head
  const int bb = wid >> 12;               // batch
  const int kvh = hh >> 2;                // GQA: GROUPS=4
  const size_t bkvS = ((size_t)bb * CKVH + kvh) * CS;

  const bf16_t* qptr = Qb + (((size_t)bb * CH + hh) * CS + qt * 16 + l16) * CHD;
  v16bf q0 = load_frag(qptr, 0, hf);
  v16bf q1 = load_frag(qptr, 32, hf);

  v8f o0 = splat8(0.f), o1 = splat8(0.f), o2 = splat8(0.f), o3 = splat8(0.f);
  v8f Mrow = splat8(-3.0e38f), Lrow = splat8(0.f);
  bf16_t* pb = &pbuf[wib * (16 * 40)];
  const float scale = 0.125f;             // 1/sqrt(64)

  const int nchunks = (qt >> 1) + 1;      // 32-key chunks covering keys <= qt*16+15
  for (int c = 0; c < nchunks; ++c) {
    const int k0 = c * 32;
    // ---- scores: two 16-key sub-tiles, K-dim = HD = 64 ----
    const bf16_t* kr0 = Kb + (bkvS + k0 + l16) * CHD;
    const bf16_t* kr1 = kr0 + 16 * CHD;
    v8f s0 = splat8(0.f), s1 = splat8(0.f);
    s0 = wmma_bf16(q0, load_frag(kr0, 0, hf), s0);
    s0 = wmma_bf16(q1, load_frag(kr0, 32, hf), s0);
    s1 = wmma_bf16(q0, load_frag(kr1, 0, hf), s1);
    s1 = wmma_bf16(q1, load_frag(kr1, 32, hf), s1);
    // ---- scale + additive causal mask (-1e9, like reference) ----
    v8f mloc;
#pragma unroll
    for (int r = 0; r < 8; ++r) {
      int row = qt * 16 + r + 8 * hf;
      float a = s0[r] * scale + ((k0 + l16) > row ? -1e9f : 0.0f);
      float b = s1[r] * scale + ((k0 + 16 + l16) > row ? -1e9f : 0.0f);
      s0[r] = a; s1[r] = b;
      mloc[r] = fmaxf(a, b);
    }
    // ---- row-max butterfly (within each 16-lane half; wave32) ----
#pragma unroll
    for (int off = 1; off < 16; off <<= 1)
#pragma unroll
      for (int r = 0; r < 8; ++r)
        mloc[r] = fmaxf(mloc[r], __shfl_xor(mloc[r], off, 32));
    v8f mnew, alpha, rs;
#pragma unroll
    for (int r = 0; r < 8; ++r) {
      mnew[r] = fmaxf(Mrow[r], mloc[r]);
      alpha[r] = __expf(Mrow[r] - mnew[r]);
      float p0 = __expf(s0[r] - mnew[r]);
      float p1 = __expf(s1[r] - mnew[r]);
      s0[r] = p0; s1[r] = p1;
      rs[r] = p0 + p1;
    }
#pragma unroll
    for (int off = 1; off < 16; off <<= 1)
#pragma unroll
      for (int r = 0; r < 8; ++r)
        rs[r] += __shfl_xor(rs[r], off, 32);
#pragma unroll
    for (int r = 0; r < 8; ++r) {
      Lrow[r] = Lrow[r] * alpha[r] + rs[r];
      Mrow[r] = mnew[r];
      o0[r] *= alpha[r]; o1[r] *= alpha[r]; o2[r] *= alpha[r]; o3[r] *= alpha[r];
    }
    // ---- P (C-layout) -> LDS -> A-layout fragment ----
#pragma unroll
    for (int r = 0; r < 8; ++r) {
      int row = r + 8 * hf;
      pb[row * 40 + l16]      = (bf16_t)s0[r];
      pb[row * 40 + 16 + l16] = (bf16_t)s1[r];
    }
    asm volatile("s_wait_dscnt 0" ::: "memory");
    v16bf pa = load_frag(pb + l16 * 40, 0, hf);
    // ---- P(16x32) x V(32x64): Vt rows are d, contiguous over s ----
    const bf16_t* vr = Vt + ((size_t)bb * CKVH + kvh) * (size_t)CHD * CS +
                       (size_t)l16 * CS;
    o0 = wmma_bf16(pa, load_frag(vr, k0, hf), o0);
    o1 = wmma_bf16(pa, load_frag(vr + (size_t)16 * CS, k0, hf), o1);
    o2 = wmma_bf16(pa, load_frag(vr + (size_t)32 * CS, k0, hf), o2);
    o3 = wmma_bf16(pa, load_frag(vr + (size_t)48 * CS, k0, hf), o3);
  }
  // ---- epilogue: normalize and emit attn_out [b*S+s, h*64+d] bf16 ----
  v8f oacc[4] = {o0, o1, o2, o3};
#pragma unroll
  for (int nt = 0; nt < 4; ++nt)
#pragma unroll
    for (int r = 0; r < 8; ++r) {
      int srow = qt * 16 + r + 8 * hf;
      int col = hh * 64 + nt * 16 + l16;
      float val = oacc[nt][r] / Lrow[r];
      Ob[((size_t)bb * CS + srow) * (size_t)CD + col] = (bf16_t)val;
    }
}

// ---------------- host launch ----------------
extern "C" void kernel_launch(void* const* d_in, const int* in_sizes, int n_in,
                              void* d_out, int out_size, void* d_ws, size_t ws_size,
                              hipStream_t stream) {
  (void)in_sizes; (void)n_in; (void)out_size; (void)ws_size;
  const float* x  = (const float*)d_in[0];
  // d_in[1] = attention_mask (causal; applied analytically in k_attn)
  const float* Wq = (const float*)d_in[2];
  const float* Wk = (const float*)d_in[3];
  const float* Wv = (const float*)d_in[4];
  const float* Wo = (const float*)d_in[5];

  float* out = (float*)d_out;                               // [B,S,D] f32
  float* kc = out + (size_t)CB * CS * CD;                   // [B,KVH,S,HD] f32
  float* vc = kc + (size_t)CB * CKVH * CS * CHD;            // [B,KVH,S,HD] f32

  const size_t NX = (size_t)CB * CS * CD;        // 8,388,608
  const size_t NWQ = (size_t)CD * CD;            // 4,194,304
  const size_t NWK = (size_t)CKVH * CHD * CD;    // 1,048,576
  const size_t NQ = (size_t)CB * CH * CS * CHD;  // 8,388,608
  const size_t NKV = (size_t)CB * CKVH * CS * CHD; // 2,097,152

  char* w = (char*)d_ws;
  bf16_t* xbf = (bf16_t*)w;  w += NX * 2;
  bf16_t* wqb = (bf16_t*)w;  w += NWQ * 2;
  bf16_t* wkb = (bf16_t*)w;  w += NWK * 2;
  bf16_t* wvb = (bf16_t*)w;  w += NWK * 2;
  bf16_t* wob = (bf16_t*)w;  w += NWQ * 2;
  bf16_t* qbf = (bf16_t*)w;  w += NQ * 2;
  bf16_t* kbf = (bf16_t*)w;  w += NKV * 2;
  bf16_t* vtb = (bf16_t*)w;  w += NKV * 2;
  bf16_t* abf = (bf16_t*)w;  w += NX * 2;        // attn output, pre-O-proj

  // f32 -> bf16 conversions (4 elems/thread)
  k_cvt<<<(int)(NX / 1024), 256, 0, stream>>>(x, xbf);
  k_cvt<<<(int)(NWQ / 1024), 256, 0, stream>>>(Wq, wqb);
  k_cvt<<<(int)(NWK / 1024), 256, 0, stream>>>(Wk, wkb);
  k_cvt<<<(int)(NWK / 1024), 256, 0, stream>>>(Wv, wvb);
  k_cvt<<<(int)(NWQ / 1024), 256, 0, stream>>>(Wo, wob);

  // projections: waves = (M/32)*(N/64); 8 waves/block
  k_proj_q<<<512, 256, 0, stream>>>(xbf, wqb, qbf);                 // 4096 waves
  k_proj_kv<<<128, 256, 0, stream>>>(xbf, wkb, kbf, kc, 0);         // 1024 waves
  k_proj_kv<<<128, 256, 0, stream>>>(xbf, wvb, vtb, vc, 1);         // 1024 waves

  // attention: one wave per (b,h,qtile) = 2*32*128 = 8192 waves
  k_attn<<<1024, 256, 0, stream>>>(qbf, kbf, vtb, abf);

  // output projection
  k_proj_o<<<512, 256, 0, stream>>>(abf, wob, out);                 // 4096 waves
}
